// RelativeSelfAttention_80281528696908
// MI455X (gfx1250) — compile-verified, matrix operand order checked by reading, not run on hardware
//
#include <hip/hip_runtime.h>

// CDNA5 (gfx1250) fused windowed multi-head attention with relative position bias.
// One 256-thread (8-wave) workgroup per WGP, persistent over WPB windows.
// All GEMMs use v_wmma_f32_16x16x32_f16 (f16 in, f32 accumulate).
// LDS (~258 KB of 320 KB/WGP) holds all weights + every intermediate for a window.

typedef __attribute__((ext_vector_type(4)))  _Float16 v4h;
typedef __attribute__((ext_vector_type(8)))  _Float16 v8h;
typedef __attribute__((ext_vector_type(16))) _Float16 v16h;
typedef __attribute__((ext_vector_type(8)))  float    v8f;

#define WIN_N 64      // tokens per window
#define CH    128     // channels
#define NH    4       // heads
#define HD    32      // head dim
#define C3    384     // 3*CH
#define NWIN  4096
#define NBLK  512
#define WPB   (NWIN / NBLK)   // windows per block
#define QSCALE 0.5f           // NUM_HEADS^-0.5

#define V8F_ZERO ((v8f){0,0,0,0,0,0,0,0})

// Load one 16-row x 32-K WMMA fragment (A role: rows are M; B role: rows are N
// when the matrix is stored transposed). Lane l, element e holds
// row = rowBase + (l&15), K = colBase + (l>=16?8:0) + e + (e>=8?8:0)
// matching the CDNA5 16-bit A/B VGPR layout. Two contiguous ds_load_b128.
__device__ __forceinline__ v16h ldfrag(const _Float16* base, int pitch,
                                       int rowBase, int colBase, int lane) {
  const _Float16* p = base + (rowBase + (lane & 15)) * pitch + colBase + ((lane >> 4) << 3);
  v8h lo = *(const v8h*)(p);
  v8h hi = *(const v8h*)(p + 16);
  return __builtin_shufflevector(lo, hi, 0,1,2,3,4,5,6,7,8,9,10,11,12,13,14,15);
}

__device__ __forceinline__ v8f wmma16(v16h a, v16h b, v8f c) {
  return __builtin_amdgcn_wmma_f32_16x16x32_f16(false, a, false, b, (short)0, c, false, false);
}

__global__ __launch_bounds__(256)
void winattn_kernel(const float* __restrict__ xg,
                    const float* __restrict__ w_qkv,  const float* __restrict__ b_qkv,
                    const float* __restrict__ w_proj, const float* __restrict__ b_proj,
                    const float* __restrict__ tbl,
                    float* __restrict__ outg) {
  __shared__ _Float16 s_wqkvT[C3 * CH];           // 96 KB, w_qkv transposed [out][in]
  __shared__ _Float16 s_wprojT[CH * CH];          // 32 KB, w_proj transposed
  __shared__ float    s_bqkv[C3];
  __shared__ float    s_bproj[CH];
  __shared__ _Float16 s_biasT[NH * WIN_N * WIN_N];// 32 KB, biasT[h][key][query]
  __shared__ _Float16 s_x[WIN_N * CH];            // 16 KB, x tile; reused as out_heads
  __shared__ _Float16 s_qk[WIN_N * 256];          // 32 KB, cols 0..127 q*scale, 128..255 k
  __shared__ _Float16 s_vT[NH * HD * WIN_N];      // 16 KB, vT[h][d][token]
  __shared__ _Float16 s_S[NH * WIN_N * WIN_N];    // 32 KB, scores then probs (in place)

  const int tid  = threadIdx.x;
  const int wave = tid >> 5;
  const int lane = tid & 31;
  const int dm   = (lane >> 4) << 3;   // D-tile row offset per lane half
  const int ln   = lane & 15;

  // ---------------- per-block staging (weights are L2-resident) ----------------
  for (int i = tid; i < (C3 * CH) / 4; i += 256) {   // float4 reads, scattered b16 writes
    float4 f = ((const float4*)w_qkv)[i];
    int base = i * 4;
    int c = base / C3, n3 = base % C3;               // C3 % 4 == 0: no row crossing
    s_wqkvT[(n3 + 0) * CH + c] = (_Float16)f.x;
    s_wqkvT[(n3 + 1) * CH + c] = (_Float16)f.y;
    s_wqkvT[(n3 + 2) * CH + c] = (_Float16)f.z;
    s_wqkvT[(n3 + 3) * CH + c] = (_Float16)f.w;
  }
  for (int i = tid; i < (CH * CH) / 4; i += 256) {
    float4 f = ((const float4*)w_proj)[i];
    int base = i * 4;
    int c = base / CH, n = base % CH;
    s_wprojT[(n + 0) * CH + c] = (_Float16)f.x;
    s_wprojT[(n + 1) * CH + c] = (_Float16)f.y;
    s_wprojT[(n + 2) * CH + c] = (_Float16)f.z;
    s_wprojT[(n + 3) * CH + c] = (_Float16)f.w;
  }
  for (int i = tid; i < C3; i += 256) s_bqkv[i]  = b_qkv[i];
  for (int i = tid; i < CH; i += 256) s_bproj[i] = b_proj[i];
  // relative position bias, transposed: biasT[h][key n][query m] = tbl[idx(m,n)][h]
  for (int i = tid; i < NH * WIN_N * WIN_N; i += 256) {
    int h = i >> 12, n = (i >> 6) & 63, m = i & 63;
    int ym = m >> 3, xm = m & 7, yn = n >> 3, xn = n & 7;
    int idx = (ym - yn + 7) * 15 + (xm - xn + 7);
    s_biasT[i] = (_Float16)tbl[idx * NH + h];
  }
  __syncthreads();

  for (int wi = 0; wi < WPB; ++wi) {
    const int win = blockIdx.x * WPB + wi;
    const size_t xoff = (size_t)win * WIN_N * CH;

    // ---- phase 1: x -> f16 LDS (float4 read, b64 packed store) ----
    {
      const float4* xv = (const float4*)(xg + xoff);
      v4h* sx4 = (v4h*)s_x;
      for (int i = tid; i < (WIN_N * CH) / 4; i += 256) {
        float4 f = xv[i];
        v4h hv = { (_Float16)f.x, (_Float16)f.y, (_Float16)f.z, (_Float16)f.w };
        sx4[i] = hv;
      }
      if (wi + 1 < WPB)   // prefetch next window's x tile (one cacheline/thread)
        __builtin_prefetch(xg + xoff + WIN_N * CH + (size_t)tid * 32, 0, 0);
    }
    __syncthreads();

    // ---- phase 2: qkv = x @ w_qkv + b (M=64,N=384,K=128) ----
    // wave owns 4 mtiles x 3 ntiles; A/B fragments shared across the tile block
    {
      const int nt0 = wave * 3;
      v8f acc[4][3];
      #pragma unroll
      for (int mt = 0; mt < 4; ++mt)
        #pragma unroll
        for (int t = 0; t < 3; ++t) acc[mt][t] = V8F_ZERO;
      #pragma unroll
      for (int ks = 0; ks < 4; ++ks) {
        v16h a[4];
        #pragma unroll
        for (int mt = 0; mt < 4; ++mt) a[mt] = ldfrag(s_x, CH, mt * 16, ks * 32, lane);
        #pragma unroll
        for (int t = 0; t < 3; ++t) {
          v16h b = ldfrag(s_wqkvT, CH, (nt0 + t) * 16, ks * 32, lane);
          #pragma unroll
          for (int mt = 0; mt < 4; ++mt) acc[mt][t] = wmma16(a[mt], b, acc[mt][t]);
        }
      }
      #pragma unroll
      for (int t = 0; t < 3; ++t) {
        const int n   = (nt0 + t) * 16 + ln;
        const float bia = s_bqkv[n];
        #pragma unroll
        for (int mt = 0; mt < 4; ++mt) {
          #pragma unroll
          for (int r = 0; r < 8; ++r) {
            const int m = mt * 16 + dm + r;
            float v = acc[mt][t][r] + bia;
            if (n < CH)          s_qk[m * 256 + n] = (_Float16)(v * QSCALE);      // q (scaled)
            else if (n < 2 * CH) s_qk[m * 256 + n] = (_Float16)v;                 // k
            else                 s_vT[(n - 2 * CH) * WIN_N + m] = (_Float16)v;    // vT[h*32+d][tok]
          }
        }
      }
    }
    __syncthreads();

    // ---- phase 3: S = q @ k^T + bias (per head, M=N=64, K=32) ----
    {
      const int h   = wave >> 1;
      const int mt0 = (wave & 1) * 2;
      v16h a0 = ldfrag(s_qk, 256, mt0 * 16,       h * HD, lane);
      v16h a1 = ldfrag(s_qk, 256, (mt0 + 1) * 16, h * HD, lane);
      #pragma unroll
      for (int nt = 0; nt < 4; ++nt) {
        v16h b = ldfrag(s_qk, 256, nt * 16, CH + h * HD, lane);  // k rows = cols of k^T
        v8f acc0 = V8F_ZERO, acc1 = V8F_ZERO;
        acc0 = wmma16(a0, b, acc0);
        acc1 = wmma16(a1, b, acc1);
        const int n = nt * 16 + ln;
        const _Float16* bt = &s_biasT[(h * WIN_N + n) * WIN_N];
        v8h bz0 = *(const v8h*)(bt + mt0 * 16 + dm);        // 8 query rows, contiguous
        v8h bz1 = *(const v8h*)(bt + (mt0 + 1) * 16 + dm);
        #pragma unroll
        for (int r = 0; r < 8; ++r) {
          const int m0 = mt0 * 16 + dm + r;
          s_S[(h * WIN_N + m0) * WIN_N + n]      = (_Float16)(acc0[r] + (float)bz0[r]);
          s_S[(h * WIN_N + m0 + 16) * WIN_N + n] = (_Float16)(acc1[r] + (float)bz1[r]);
        }
      }
    }
    __syncthreads();

    // ---- phase 4: row softmax, one row per thread, vectorized in-place f16 ----
    {
      const int h = tid >> 6, r = tid & 63;
      _Float16* p = &s_S[(h * WIN_N + r) * WIN_N];
      float vals[WIN_N];
      #pragma unroll
      for (int jb = 0; jb < 8; ++jb) {
        v8h vv = *(const v8h*)(p + jb * 8);
        #pragma unroll
        for (int e = 0; e < 8; ++e) vals[jb * 8 + e] = (float)vv[e];
      }
      float mx = -3.0e38f;
      #pragma unroll
      for (int j = 0; j < WIN_N; ++j) mx = fmaxf(mx, vals[j]);
      float sum = 0.f;
      #pragma unroll
      for (int j = 0; j < WIN_N; ++j) { float e = __expf(vals[j] - mx); vals[j] = e; sum += e; }
      const float inv = 1.0f / sum;
      #pragma unroll
      for (int jb = 0; jb < 8; ++jb) {
        v8h vv;
        #pragma unroll
        for (int e = 0; e < 8; ++e) vv[e] = (_Float16)(vals[jb * 8 + e] * inv);
        *(v8h*)(p + jb * 8) = vv;
      }
    }
    __syncthreads();

    // ---- phase 5: O_h = P @ V (per head, M=64, N=32, K=64) -> s_x (out_heads) ----
    {
      const int h   = wave >> 1;
      const int mt0 = (wave & 1) * 2;
      const _Float16* Sh = &s_S[h * WIN_N * WIN_N];
      const _Float16* Vh = &s_vT[h * HD * WIN_N];
      v8f acc[2][2];
      #pragma unroll
      for (int i = 0; i < 2; ++i)
        #pragma unroll
        for (int t = 0; t < 2; ++t) acc[i][t] = V8F_ZERO;
      #pragma unroll
      for (int ks = 0; ks < 2; ++ks) {
        v16h a0 = ldfrag(Sh, WIN_N, mt0 * 16,       ks * 32, lane);
        v16h a1 = ldfrag(Sh, WIN_N, (mt0 + 1) * 16, ks * 32, lane);
        v16h b0 = ldfrag(Vh, WIN_N, 0,  ks * 32, lane);   // vT rows = d
        v16h b1 = ldfrag(Vh, WIN_N, 16, ks * 32, lane);
        acc[0][0] = wmma16(a0, b0, acc[0][0]);
        acc[0][1] = wmma16(a0, b1, acc[0][1]);
        acc[1][0] = wmma16(a1, b0, acc[1][0]);
        acc[1][1] = wmma16(a1, b1, acc[1][1]);
      }
      #pragma unroll
      for (int i = 0; i < 2; ++i) {
        #pragma unroll
        for (int t = 0; t < 2; ++t) {
          const int n = h * HD + t * 16 + ln;
          #pragma unroll
          for (int r = 0; r < 8; ++r) {
            const int m = (mt0 + i) * 16 + dm + r;
            s_x[m * CH + n] = (_Float16)acc[i][t][r];     // x buffer reused as out_heads
          }
        }
      }
    }
    __syncthreads();

    // ---- phase 6: out = out_heads @ w_proj + b (M=64,N=128,K=128) -> global f32 ----
    {
      const int mt0 = (wave >> 2) * 2;   // {0,2}
      const int ntb = (wave & 3) * 2;    // {0,2,4,6}
      v8f acc[2][2];
      #pragma unroll
      for (int i = 0; i < 2; ++i)
        #pragma unroll
        for (int t = 0; t < 2; ++t) acc[i][t] = V8F_ZERO;
      #pragma unroll
      for (int ks = 0; ks < 4; ++ks) {
        v16h a0 = ldfrag(s_x, CH, mt0 * 16,       ks * 32, lane);
        v16h a1 = ldfrag(s_x, CH, (mt0 + 1) * 16, ks * 32, lane);
        v16h b0 = ldfrag(s_wprojT, CH, ntb * 16,       ks * 32, lane);
        v16h b1 = ldfrag(s_wprojT, CH, (ntb + 1) * 16, ks * 32, lane);
        acc[0][0] = wmma16(a0, b0, acc[0][0]);
        acc[0][1] = wmma16(a0, b1, acc[0][1]);
        acc[1][0] = wmma16(a1, b0, acc[1][0]);
        acc[1][1] = wmma16(a1, b1, acc[1][1]);
      }
      // single per-lane base; all 32 stores use compile-time offsets
      float* op = outg + xoff + (size_t)((mt0 * 16 + dm) * CH + ntb * 16 + ln);
      const float bia0 = s_bproj[ntb * 16 + ln];
      const float bia1 = s_bproj[ntb * 16 + 16 + ln];
      #pragma unroll
      for (int i = 0; i < 2; ++i) {
        #pragma unroll
        for (int r = 0; r < 8; ++r) {
          op[(i * 16 + r) * CH]      = acc[i][0][r] + bia0;
          op[(i * 16 + r) * CH + 16] = acc[i][1][r] + bia1;
        }
      }
    }
    __syncthreads();   // before next window reuses LDS
  }
}

extern "C" void kernel_launch(void* const* d_in, const int* in_sizes, int n_in,
                              void* d_out, int out_size, void* d_ws, size_t ws_size,
                              hipStream_t stream) {
  (void)in_sizes; (void)n_in; (void)d_ws; (void)ws_size; (void)out_size;
  const float* x      = (const float*)d_in[0];
  const float* w_qkv  = (const float*)d_in[1];
  const float* b_qkv  = (const float*)d_in[2];
  const float* w_proj = (const float*)d_in[3];
  const float* b_proj = (const float*)d_in[4];
  const float* tbl    = (const float*)d_in[5];
  float* out = (float*)d_out;
  winattn_kernel<<<NBLK, 256, 0, stream>>>(x, w_qkv, b_qkv, w_proj, b_proj, tbl, out);
}
